// WindowAttention_57878979280983
// MI455X (gfx1250) — matrix operand order repaired and checked by reading, not compiled, not run on hardware
//
#include <hip/hip_runtime.h>
#include <hip/hip_bf16.h>

typedef __attribute__((ext_vector_type(16))) _Float16 v16h;
typedef __attribute__((ext_vector_type(8)))  float    v8f;

#define DIM   384
#define HEADS 12
#define HD    32
#define NTOK  49
#define NPAD  64
#define QK_SCALE 0.17677669529663687f  // 1/sqrt(32)

#define QKVW_ELEMS (3 * DIM * DIM)     // 442368
#define PRJW_ELEMS (DIM * DIM)         // 147456

union FragU { v16h v; uint4 u[2]; };
union H8   { uint4 u; _Float16 h[8]; };

// A-fragment (16x32 f16, MxK) from row-major [row][K] f16 storage (LDS or global).
// lane: row = rowb + (lane&15); elem i<8 -> K = kbase + (lane>>4)*8 + i
//                               elem i>=8 -> K = kbase + 16 + (lane>>4)*8 + (i-8)
__device__ inline v16h load_a_h16(const _Float16* base, int ld, int rowb, int kbase, int lane) {
  const _Float16* p = base + (rowb + (lane & 15)) * ld + kbase + ((lane >> 4) * 8);
  FragU f;
  f.u[0] = *(const uint4*)p;
  f.u[1] = *(const uint4*)(p + 16);
  return f.v;
}

// B-fragment (32x16 f16, KxN) from N-major (row n, K contiguous) f16 storage.
// lane: col = rowb + (lane&15); elem i<8 -> K = kbase + (lane>>4)*8 + i
//                               elem i>=8 -> K = kbase + 16 + (lane>>4)*8 + (i-8)
__device__ inline v16h load_b_h16(const _Float16* base, int ld, int rowb, int kbase, int lane) {
  const _Float16* p = base + (rowb + (lane & 15)) * ld + kbase + ((lane >> 4) * 8);
  FragU f;
  f.u[0] = *(const uint4*)p;
  f.u[1] = *(const uint4*)(p + 16);
  return f.v;
}

// ---- one-shot weight fp32 -> f16 conversion into workspace ----
__global__ __launch_bounds__(256)
void convert_weights(const float* __restrict__ qkv_w,
                     const float* __restrict__ proj_w,
                     _Float16* __restrict__ wh)   // [QKVW_ELEMS + PRJW_ELEMS] f16
{
  const int nq4 = QKVW_ELEMS / 4;
  const int nt4 = (QKVW_ELEMS + PRJW_ELEMS) / 4;
  for (int j = blockIdx.x * blockDim.x + threadIdx.x; j < nt4; j += gridDim.x * blockDim.x) {
    float4 v = (j < nq4) ? *(const float4*)(qkv_w + (size_t)j * 4)
                         : *(const float4*)(proj_w + (size_t)(j - nq4) * 4);
    H8 t;
    t.h[0]=(_Float16)v.x; t.h[1]=(_Float16)v.y; t.h[2]=(_Float16)v.z; t.h[3]=(_Float16)v.w;
    *(uint2*)(wh + (size_t)j * 4) = make_uint2(t.u.x, t.u.y);
  }
}

__global__ __launch_bounds__(256)
void winattn_fused(const float* __restrict__ x,
                   const _Float16* __restrict__ qkv_wh,   // f16 [1152][384]
                   const float* __restrict__ qkv_b,
                   const _Float16* __restrict__ proj_wh,  // f16 [384][384]
                   const float* __restrict__ proj_b,
                   const float* __restrict__ rpb_table,
                   const int*   __restrict__ rel_index,
                   float* __restrict__ out)
{
  extern __shared__ char smem[];
  _Float16* xs  = (_Float16*)smem;                 // [64][384] f16; reused as attn-out [m][c]
  _Float16* qs  = xs + NPAD * DIM;                 // [12][64][32] q (pre-scaled)
  _Float16* ks  = qs + HEADS * NPAD * HD;          // [12][64][32] k
  _Float16* vT  = ks + HEADS * NPAD * HD;          // [12][32][64] v transposed [d][n]
  _Float16* ps  = vT + HEADS * HD * NPAD;          // [8 waves][16][64] P scratch [m][n]
  float*    tab = (float*)(ps + 8 * 16 * 64);      // [169*12]
  int*      rel = (int*)(tab + 169 * HEADS);       // [49*49], pre-scaled by HEADS

  const int tid  = threadIdx.x;
  const int lane = tid & 31;
  const int wave = tid >> 5;
  const int b    = blockIdx.x;
  const int crow = (lane >> 4) * 8;  // C-fragment row base (contiguous dim)
  const int ccol = lane & 15;        // C-fragment column

  for (int i = tid; i < 169 * HEADS; i += 256) tab[i] = rpb_table[i];
  for (int i = tid; i < NTOK * NTOK; i += 256) rel[i] = rel_index[i] * HEADS;

  // ---- x window -> LDS f16, rows >= 49 zeroed
  const float* xg = x + (size_t)b * NTOK * DIM;
  for (int i = tid * 4; i < NPAD * DIM; i += 256 * 4) {
    int row = i / DIM, col = i % DIM;
    float4 v = (row < NTOK) ? *(const float4*)(xg + row * DIM + col)
                            : make_float4(0.f, 0.f, 0.f, 0.f);
    H8 t;  // only 4 used
    t.h[0]=(_Float16)v.x; t.h[1]=(_Float16)v.y; t.h[2]=(_Float16)v.z; t.h[3]=(_Float16)v.w;
    *(uint2*)(xs + i) = make_uint2(t.u.x, t.u.y);
  }
  __syncthreads();

  // ====== QKV, transposed orientation: qkvT[j,m] = sum_c W[j,c] * x[m,c] ======
  // C layout: lane -> col m = mt*16+ccol (fixed), rows j = jt*16+crow+r (contiguous)
  for (int jt = wave; jt < 72; jt += 8) {
    int jbase = jt * 16;
    v8f acc[4] = {};
    for (int kc = 0; kc < 12; ++kc) {
      v16h af = load_a_h16(qkv_wh, DIM, jbase, kc * 32, lane);
#pragma unroll
      for (int mt = 0; mt < 4; ++mt) {
        v16h bf = load_b_h16(xs, DIM, mt * 16, kc * 32, lane);
        acc[mt] = __builtin_amdgcn_wmma_f32_16x16x32_f16(
            false, af, false, bf, (short)0, acc[mt], false, false);
      }
    }
    int j0 = jbase + crow;               // first of 8 contiguous j per lane
    float4 qb0 = *(const float4*)(qkv_b + j0);
    float4 qb1 = *(const float4*)(qkv_b + j0 + 4);
    float qb[8] = {qb0.x,qb0.y,qb0.z,qb0.w,qb1.x,qb1.y,qb1.z,qb1.w};
    if (jt < 24) {                                        // ---- Q (scaled)
      int h = j0 >> 5, d0 = j0 & 31;
#pragma unroll
      for (int mt = 0; mt < 4; ++mt) {
        int m = mt * 16 + ccol;
        H8 t;
#pragma unroll
        for (int r = 0; r < 8; ++r) t.h[r] = (_Float16)((acc[mt][r] + qb[r]) * QK_SCALE);
        *(uint4*)(qs + h * NPAD * HD + m * HD + d0) = t.u;
      }
    } else if (jt < 48) {                                 // ---- K
      int jj = j0 - DIM, h = jj >> 5, d0 = jj & 31;
#pragma unroll
      for (int mt = 0; mt < 4; ++mt) {
        int m = mt * 16 + ccol;
        H8 t;
#pragma unroll
        for (int r = 0; r < 8; ++r) t.h[r] = (_Float16)(acc[mt][r] + qb[r]);
        *(uint4*)(ks + h * NPAD * HD + m * HD + d0) = t.u;
      }
    } else {                                              // ---- V -> [d][n] transposed
      int jj = j0 - 2 * DIM, h = jj >> 5, d0 = jj & 31;
#pragma unroll
      for (int mt = 0; mt < 4; ++mt) {
        int m = mt * 16 + ccol;
#pragma unroll
        for (int r = 0; r < 8; ++r)
          vT[h * HD * NPAD + (d0 + r) * NPAD + m] = (_Float16)(acc[mt][r] + qb[r]);
      }
    }
  }
  __syncthreads();  // q/k/vT ready; xs dead -> becomes attn output [m][c]

  // ====== Attention, transposed: T[n,m] = S[m,n]; softmax over rows of T ======
  _Float16* pm = ps + wave * 16 * 64;   // per-wave P scratch, row-major [m_local][n]
  for (int h = wave; h < HEADS; h += 8) {
    const _Float16* qh = qs + h * NPAD * HD;
    const _Float16* kh = ks + h * NPAD * HD;
    const _Float16* vh = vT + h * HD * NPAD;
    v16h kA[4], qB[4], vA[2][2];
#pragma unroll
    for (int nt = 0; nt < 4; ++nt) kA[nt] = load_a_h16(kh, HD, nt * 16, 0, lane);
#pragma unroll
    for (int mtb = 0; mtb < 4; ++mtb) qB[mtb] = load_b_h16(qh, HD, mtb * 16, 0, lane);
#pragma unroll
    for (int dt = 0; dt < 2; ++dt)
#pragma unroll
      for (int kc = 0; kc < 2; ++kc)
        vA[dt][kc] = load_a_h16(vh, NPAD, dt * 16, kc * 32, lane);

    for (int mtb = 0; mtb < 4; ++mtb) {     // tile of 16 query tokens (columns of T)
      int m = mtb * 16 + ccol;              // query token for this lane
      v8f t[4];
#pragma unroll
      for (int nt = 0; nt < 4; ++nt) {
        v8f z = {};
        t[nt] = __builtin_amdgcn_wmma_f32_16x16x32_f16(
            false, kA[nt], false, qB[mtb], (short)0, z, false, false);
      }
      // bias + key mask (rows n of T)
#pragma unroll
      for (int nt = 0; nt < 4; ++nt) {
#pragma unroll
        for (int r = 0; r < 8; ++r) {
          int n = nt * 16 + crow + r;
          if (n >= NTOK)     t[nt][r] = -1e30f;
          else if (m < NTOK) t[nt][r] += tab[rel[m * NTOK + n] + h];
        }
      }
      // softmax over n: 32 in-lane values + one cross-half-wave exchange
      float mx = -1e30f;
#pragma unroll
      for (int nt = 0; nt < 4; ++nt)
#pragma unroll
        for (int r = 0; r < 8; ++r) mx = fmaxf(mx, t[nt][r]);
      mx = fmaxf(mx, __shfl_xor(mx, 16, 32));
      float sum = 0.f;
#pragma unroll
      for (int nt = 0; nt < 4; ++nt)
#pragma unroll
        for (int r = 0; r < 8; ++r) { t[nt][r] = __expf(t[nt][r] - mx); sum += t[nt][r]; }
      sum += __shfl_xor(sum, 16, 32);
      float inv = 1.0f / (sum + 1e-20f);
      // store P[m_local][n] : 8 contiguous n per lane -> b128 stores
#pragma unroll
      for (int nt = 0; nt < 4; ++nt) {
        H8 w;
#pragma unroll
        for (int r = 0; r < 8; ++r) w.h[r] = (_Float16)(t[nt][r] * inv);
        *(uint4*)(pm + ccol * 64 + nt * 16 + crow) = w.u;
      }
      // O^T[d,m] = sum_n V^T[d,n] * P^T[n,m]  (A = vT, B = P rows)
#pragma unroll
      for (int dt = 0; dt < 2; ++dt) {
        v8f o = {};
#pragma unroll
        for (int kc = 0; kc < 2; ++kc) {
          v16h pb = load_b_h16(pm, 64, 0, kc * 32, lane);
          o = __builtin_amdgcn_wmma_f32_16x16x32_f16(
              false, vA[dt][kc], false, pb, (short)0, o, false, false);
        }
        // lane: token m fixed, 8 contiguous d -> b128 store into row-major attn-out
        H8 w;
#pragma unroll
        for (int r = 0; r < 8; ++r) w.h[r] = (_Float16)o[r];
        *(uint4*)(xs + m * DIM + h * HD + dt * 16 + crow) = w.u;
      }
    }
  }
  __syncthreads();

  // ====== Projection, transposed: outT[j,m] = sum_c W[j,c] * y[m,c] ======
  float* og = out + (size_t)b * NTOK * DIM;
  for (int tix = wave; tix < 96; tix += 8) {
    int mt = tix & 3, jt = tix >> 2;
    int jbase = jt * 16;
    v8f acc = {};
    for (int kc = 0; kc < 12; ++kc) {
      v16h af = load_a_h16(proj_wh, DIM, jbase, kc * 32, lane);
      v16h bf = load_b_h16(xs, DIM, mt * 16, kc * 32, lane);
      acc = __builtin_amdgcn_wmma_f32_16x16x32_f16(
          false, af, false, bf, (short)0, acc, false, false);
    }
    int m = mt * 16 + ccol;              // output row (fixed per lane)
    int j0 = jbase + crow;               // 8 contiguous output cols
    float4 pb0 = *(const float4*)(proj_b + j0);
    float4 pb1 = *(const float4*)(proj_b + j0 + 4);
    if (m < NTOK) {
      float4 o0 = make_float4(acc[0]+pb0.x, acc[1]+pb0.y, acc[2]+pb0.z, acc[3]+pb0.w);
      float4 o1 = make_float4(acc[4]+pb1.x, acc[5]+pb1.y, acc[6]+pb1.z, acc[7]+pb1.w);
      *(float4*)(og + m * DIM + j0)     = o0;
      *(float4*)(og + m * DIM + j0 + 4) = o1;
    }
  }
}

extern "C" void kernel_launch(void* const* d_in, const int* in_sizes, int n_in,
                              void* d_out, int out_size, void* d_ws, size_t ws_size,
                              hipStream_t stream) {
  (void)in_sizes; (void)n_in; (void)out_size; (void)ws_size;
  const float* x     = (const float*)d_in[0];
  const float* qkv_w = (const float*)d_in[1];
  const float* qkv_b = (const float*)d_in[2];
  const float* prj_w = (const float*)d_in[3];
  const float* prj_b = (const float*)d_in[4];
  const float* rpb   = (const float*)d_in[5];
  const int*   rel   = (const int*)d_in[6];
  float* outp = (float*)d_out;

  _Float16* wh = (_Float16*)d_ws;                  // [QKVW_ELEMS + PRJW_ELEMS] f16 (~1.2 MB)
  const _Float16* qkv_wh = wh;
  const _Float16* prj_wh = wh + QKVW_ELEMS;

  // stage 1: weights fp32 -> f16 (runs every call; deterministic, capture-safe)
  convert_weights<<<576, 256, 0, stream>>>(qkv_w, prj_w, wh);

  // stage 2: fused window attention
  size_t smem = (size_t)(NPAD * DIM + 3 * HEADS * NPAD * HD + 8 * 16 * 64) * 2
              + (size_t)(169 * HEADS) * 4 + (size_t)(NTOK * NTOK) * 4;  // ~231 KB

  hipFuncSetAttribute((const void*)winattn_fused,
                      hipFuncAttributeMaxDynamicSharedMemorySize, (int)smem);
  winattn_fused<<<4096, 256, smem, stream>>>(x, qkv_wh, qkv_b, prj_wh, prj_b, rpb, rel, outp);
}
